// Scaled_Dot_Product_Attention_neg_28389733826703
// MI455X (gfx1250) — matrix operand order repaired and checked by reading, not compiled, not run on hardware
//
#include <hip/hip_runtime.h>
#include <math.h>

// Scaled_Dot_Product_Attention_neg — CDNA5 (gfx1250) WMMA implementation.
// B=16, LQ=LK=2048, D=128, fp32 in/out. bf16 hi/lo split (3-term) WMMA for
// near-fp32 accuracy at matrix-core throughput. Phase-3 is software-pipelined
// (register double-buffer of the kg_sim/V stream) since the 268MB kg_sim
// stream sets the ~14us HBM floor on MI455X (23.3 TB/s).

#define NB 16
#define SLQ 2048
#define SLK 2048
#define DD 128
#define QT 16            // query rows per workgroup
#define NTHR 256         // 8 wave32
#define NW 8
#define SC_STRIDE 2052   // scores row stride (f32), staggers LDS banks, mult of 4
#define V_STRIDE 132     // V staging row stride (f32), mult of 4 for b128 stores

typedef __attribute__((ext_vector_type(16))) __bf16 v16bf;
typedef __attribute__((ext_vector_type(8)))  float  v8f;
typedef __attribute__((ext_vector_type(4)))  float  f32x4;

static __device__ __forceinline__ v8f wmma_bf16(v16bf a, v16bf b, v8f c) {
  // (neg_a, A, neg_b, B, c_mod, C, reuse_a, reuse_b)
  return __builtin_amdgcn_wmma_f32_16x16x32_bf16(false, a, false, b, (short)0, c,
                                                 false, false);
}

static __device__ __forceinline__ void split2(float x, __bf16 &hi, __bf16 &lo) {
  __bf16 h = (__bf16)x;
  hi = h;
  lo = (__bf16)(x - (float)h);
}

static __device__ __forceinline__ unsigned pack_bf16x2(float e, float o) {
  __bf16 a = (__bf16)e, b = (__bf16)o;   // [15:0]=even-k elem, [31:16]=odd-k elem
  return ((unsigned)__builtin_bit_cast(unsigned short, b) << 16) |
         (unsigned)__builtin_bit_cast(unsigned short, a);
}

__global__ __launch_bounds__(NTHR)
void attn_neg_kernel(const float* __restrict__ Q, const float* __restrict__ K,
                     const float* __restrict__ V, const float* __restrict__ scale_p,
                     const float* __restrict__ kg, float* __restrict__ out) {
  extern __shared__ float smem[];
  float*    scores = smem;                          // QT * SC_STRIDE (~128 KB)
  float*    vs     = scores + QT * SC_STRIDE;       // 32 * V_STRIDE  (~16.5 KB)
  unsigned* awh    = (unsigned*)(vs + 32*V_STRIDE); // 32 lanes * 8 dwords (W hi operand)
  unsigned* awl    = awh + 32*8;                    // 32 lanes * 8 dwords (W lo operand)
  float*    red    = (float*)(awl + 32*8);          // NTHR
  float*    mrow   = red + NTHR;                    // QT
  float*    invZ   = mrow + QT;                     // QT
  float*    dinv   = invZ + QT;                     // QT

  const int tid  = threadIdx.x;
  const int lane = tid & 31;
  const int wid  = tid >> 5;
  const int col  = lane & 15;   // A-row / B-col / D-col index for this lane
  const int half = lane >> 4;

  const int b  = blockIdx.x >> 7;          // 128 q-tiles per batch
  const int q0 = (blockIdx.x & 127) * QT;

  const float scale = scale_p[0];

  // ---------- Phase 1: scores = -(Q K^T) * scale via 3-term bf16 WMMA ----------
  // A operand (Q tile, 16x32 per chunk): lane holds row M=col;
  // element e maps to K = half*16 + e (contiguous, matches B).
  v16bf qhi[4], qlo[4];
  {
    const float* qrow = Q + ((size_t)b * SLQ + q0 + col) * DD;
    #pragma unroll
    for (int c = 0; c < 4; ++c) {
      #pragma unroll
      for (int e = 0; e < 16; ++e) {
        int kA = c*32 + half*16 + e;
        __bf16 h, l; split2(qrow[kA], h, l);
        qhi[c][e] = h; qlo[c][e] = l;
      }
    }
  }

  for (int kt = wid; kt < SLK/16; kt += NW) {
    // B operand = K^T chunk: lane holds column col (= a K row), element e maps
    // to reduction index half*16 + e -> contiguous 16 floats along d.
    const float* krow = K + ((size_t)b * SLK + kt*16 + col) * DD;
    v8f acc = {};
    #pragma unroll
    for (int c = 0; c < 4; ++c) {
      v16bf bhi, blo;
      #pragma unroll
      for (int e = 0; e < 16; ++e) {
        __bf16 h, l; split2(krow[c*32 + half*16 + e], h, l);
        bhi[e] = h; blo[e] = l;
      }
      acc = wmma_bf16(qhi[c], bhi, acc);
      acc = wmma_bf16(qlo[c], bhi, acc);
      acc = wmma_bf16(qhi[c], blo, acc);
    }
    // D layout: VGPR i <-> row i + 8*half, column col
    #pragma unroll
    for (int i = 0; i < 8; ++i) {
      int row = i + 8*half;
      scores[row * SC_STRIDE + kt*16 + col] = -acc[i] * scale;
    }
  }
  __syncthreads();

  // ---------- Phase 2: first softmax: scores <- exp(s - m); invZ[row] ----------
  {
    const int r = tid >> 4, c0 = tid & 15;
    float m = -INFINITY;
    for (int j = 0; j < SLK/16; ++j)
      m = fmaxf(m, scores[r*SC_STRIDE + c0 + 16*j]);
    red[tid] = m;
    __syncthreads();
    if (tid < QT) {
      float mm = -INFINITY;
      #pragma unroll
      for (int j = 0; j < 16; ++j) mm = fmaxf(mm, red[tid*16 + j]);
      mrow[tid] = mm;
    }
    __syncthreads();
    const float mr = mrow[r];
    float z = 0.f;
    for (int j = 0; j < SLK/16; ++j) {
      int idx = r*SC_STRIDE + c0 + 16*j;
      float p = __expf(scores[idx] - mr);
      scores[idx] = p;
      z += p;
    }
    red[tid] = z;
    __syncthreads();
    if (tid < QT) {
      float s = 0.f;
      #pragma unroll
      for (int j = 0; j < 16; ++j) s += red[tid*16 + j];
      invZ[tid] = 1.f / s;
    }
    __syncthreads();
  }

  // ---------- Phase 3: w = exp(-softmax * kg); ctx = W @ V; den = rowsum(W) ----
  // Second softmax logits lie in (-1, 0] -> no max subtraction needed (exact).
  // Software pipeline: global loads of chunk kc+1 (kg pair + V float4s) are
  // issued into registers right after staging chunk kc, and land while the
  // WMMAs of chunk kc execute.
  const int qw = tid >> 4;              // this thread's W row (fixed)
  const int kk = (tid & 15) * 2;        // this thread's even/odd k pair
  const float invZq = invZ[qw];
  const float* kgrow = kg + ((size_t)b * SLQ + q0 + qw) * (size_t)SLK + kk;
  const float* vb = V + (size_t)b * SLK * DD;
  // LDS slot of this thread's packed A-operand dword (ISA 16-bit A layout):
  // lane = half*16 + row, dword v holds k pair (even,odd).
  const int awIdx = ((((kk >> 3) & 1) * 16 + qw) * 8) +
                    ((kk & 7) >> 1) + ((kk >> 4) << 2);

  float den_local = 0.f;
  v8f ctx = {};                         // this wave's 16x16 output tile (fp32)
  const int dtile = wid;                // wave owns output columns [16w, 16w+16)

  float g0, g1;                         // pipelined kg pair
  f32x4 vreg[4];                        // pipelined V chunk slice (16 floats)
  {
    g0 = kgrow[0]; g1 = kgrow[1];
    #pragma unroll
    for (int rnd = 0; rnd < 4; ++rnd) {
      int f = rnd * NTHR + tid;
      vreg[rnd] = *(const f32x4*)(vb + (size_t)(f >> 5) * DD + (f & 31) * 4);
    }
  }

  for (int kc = 0; kc < SLK/32; ++kc) {
    __syncthreads();   // all waves done reading LDS of previous chunk
    // --- stage chunk kc from registers into LDS ---
    {
      float s1a = scores[qw*SC_STRIDE + kc*32 + kk]     * invZq;
      float s1b = scores[qw*SC_STRIDE + kc*32 + kk + 1] * invZq;
      float w0 = __expf(-s1a * g0);
      float w1 = __expf(-s1b * g1);
      den_local += w0 + w1;
      awh[awIdx] = pack_bf16x2(w0, w1);
      __bf16 t0 = (__bf16)w0, t1 = (__bf16)w1;
      awl[awIdx] = pack_bf16x2(w0 - (float)t0, w1 - (float)t1);
      #pragma unroll
      for (int rnd = 0; rnd < 4; ++rnd) {
        int f = rnd * NTHR + tid;
        *(f32x4*)(vs + (f >> 5) * V_STRIDE + (f & 31) * 4) = vreg[rnd];
      }
    }
    // --- issue next chunk's global loads (overlap with WMMA below) ---
    if (kc + 1 < SLK/32) {
      g0 = kgrow[(size_t)(kc+1)*32];
      g1 = kgrow[(size_t)(kc+1)*32 + 1];
      const float* vchunk = vb + (size_t)((kc+1)*32) * DD;
      #pragma unroll
      for (int rnd = 0; rnd < 4; ++rnd) {
        int f = rnd * NTHR + tid;
        vreg[rnd] = *(const f32x4*)(vchunk + (f >> 5) * DD + (f & 31) * 4);
      }
    }
    __syncthreads();
    // --- per-wave WMMA: ctx(16x16) += W(16x32) x V(32x16), 3-term split ---
    const v16bf ahi = *(const v16bf*)(awh + lane*8);  // 2x ds b128, zero VALU
    const v16bf alo = *(const v16bf*)(awl + lane*8);
    v16bf bhi, blo;
    #pragma unroll
    for (int e = 0; e < 16; ++e) {
      __bf16 h, l; split2(vs[((e & 8) * 2 + half*8 + (e & 7)) * V_STRIDE + dtile*16 + col], h, l);
      bhi[e] = h; blo[e] = l;
    }
    ctx = wmma_bf16(ahi, bhi, ctx);
    ctx = wmma_bf16(alo, bhi, ctx);
    ctx = wmma_bf16(ahi, blo, ctx);
  }

  // denominator reduction and normalized output
  red[tid] = den_local;
  __syncthreads();
  if (tid < QT) {
    float s = 0.f;
    #pragma unroll
    for (int j = 0; j < 16; ++j) s += red[tid*16 + j];
    dinv[tid] = 1.f / s;
  }
  __syncthreads();
  #pragma unroll
  for (int i = 0; i < 8; ++i) {
    int row = i + 8*half;
    out[((size_t)b * SLQ + q0 + row) * DD + dtile*16 + col] = ctx[i] * dinv[row];
  }
}

extern "C" void kernel_launch(void* const* d_in, const int* in_sizes, int n_in,
                              void* d_out, int out_size, void* d_ws, size_t ws_size,
                              hipStream_t stream) {
  (void)in_sizes; (void)n_in; (void)out_size; (void)d_ws; (void)ws_size;
  const float* Q     = (const float*)d_in[0];
  const float* K     = (const float*)d_in[1];
  const float* V     = (const float*)d_in[2];
  const float* scale = (const float*)d_in[3];
  const float* kg    = (const float*)d_in[4];
  float* out = (float*)d_out;

  const size_t smem = (size_t)(QT*SC_STRIDE + 32*V_STRIDE) * sizeof(float)
                    + (size_t)(2 * 32 * 8) * sizeof(unsigned)
                    + (size_t)(NTHR + 3*QT) * sizeof(float);  // ~151.5 KB (CDNA5: 320 KB LDS/WGP)
  (void)hipFuncSetAttribute((const void*)attn_neg_kernel,
                            hipFuncAttributeMaxDynamicSharedMemorySize, (int)smem);

  dim3 grid(NB * (SLQ / QT));   // 2048 workgroups
  dim3 block(NTHR);             // 8 wave32
  attn_neg_kernel<<<grid, block, smem, stream>>>(Q, K, V, scale, kg, out);
}
